// GRUModel_55654186222262
// MI455X (gfx1250) — compile-verified, matrix operand order checked by reading, not run on hardware
//
#include <hip/hip_runtime.h>

// ---------------------------------------------------------------------------
// Model dims (fixed by the reference)
// ---------------------------------------------------------------------------
#define VV   10000   // vocab
#define EE   512     // embed dim
#define HH   1024    // hidden
#define BB   64      // batch
#define TT   64      // encoder steps
#define DEC  179     // decoder steps

typedef __attribute__((ext_vector_type(16))) __bf16 v16bf;
typedef __attribute__((ext_vector_type(8)))  float  v8f;

typedef unsigned short u16;

// ---------------------------------------------------------------------------
// bf16 helpers (round-to-nearest-even f32 -> bf16)
// ---------------------------------------------------------------------------
__device__ __forceinline__ u16 f2bf(float f) {
    union { float f; unsigned int u; } a; a.f = f;
    unsigned int u = a.u;
    u += 0x7fffu + ((u >> 16) & 1u);
    return (u16)(u >> 16);
}
__device__ __forceinline__ float bf2f(u16 h) {
    union { unsigned int u; float f; } a; a.u = ((unsigned int)h) << 16;
    return a.f;
}
__device__ __forceinline__ float sigmoidf(float x) {
    return 1.0f / (1.0f + __expf(-x));
}

// ---------------------------------------------------------------------------
// CDNA5 async copy: global -> LDS, 16 bytes per lane, tracked on ASYNCcnt.
//   global_load_async_to_lds_b128  vdst(lds addr), vaddr(64b global), off
// ---------------------------------------------------------------------------
__device__ __forceinline__ void async_cp16(u16* lds_dst, const u16* gsrc) {
    unsigned lds_off = (unsigned)(unsigned long long)lds_dst; // addr[31:0] = LDS offset
    asm volatile("global_load_async_to_lds_b128 %0, %1, off"
                 :: "v"(lds_off), "v"(gsrc) : "memory");
}
__device__ __forceinline__ void wait_async0() {
    asm volatile("s_wait_asynccnt 0x0" ::: "memory");
}

// ---------------------------------------------------------------------------
// WMMA fragment loads.
// 16-bit A-matrix 16x32 ISA layout: lanes 0-15 hold K=[0..7],[16..23],
// lanes 16-31 hold K=[8..15],[24..31].  Per lane: two contiguous 16B chunks
// separated by 16 elements -> two b128 loads (global or ds).
// ---------------------------------------------------------------------------
__device__ __forceinline__ v16bf ldfrag(const u16* p) {
    uint4 a = *reinterpret_cast<const uint4*>(p);
    uint4 b = *reinterpret_cast<const uint4*>(p + 16);
    union { uint4 u[2]; v16bf v; } f;
    f.u[0] = a; f.u[1] = b;
    return f.v;
}
__device__ __forceinline__ v16bf load_frag(const u16* base, int row, int ld,
                                           int k0, int hi) {
    return ldfrag(base + (size_t)row * ld + k0 + hi * 8);
}

// ---------------------------------------------------------------------------
// Fused GRU step:  h' = (1-z)*n + z*h
// Grid: HH/64 = 16 blocks x 256 threads (8 waves). Each WG: 64x64 tile of h'.
// Activations (x: 64xKx, h: 64xHH) are staged into LDS with async copies
// (rows padded +8 elems = 4 dwords to spread banks); weights stream from
// global (L2-resident bf16). Each wave: two 16x16 subtiles, 4 accumulators
// (r, z combined gi+gh; i_n and h_n separate since r scales only h_n).
// ---------------------------------------------------------------------------
template <int Kx>
__global__ __launch_bounds__(256) void gru_step(
    const u16* __restrict__ x,
    const u16* __restrict__ h,
    const u16* __restrict__ Wih, const u16* __restrict__ Whh,
    const float* __restrict__ bih, const float* __restrict__ bhh,
    u16* __restrict__ hout)
{
    constexpr int KP = Kx + 8;    // padded LDS row (bf16 elems)
    constexpr int HP = HH + 8;
    extern __shared__ u16 smem[];
    u16* sx = smem;               // 64 x KP
    u16* shm = smem + 64 * KP;    // 64 x HP

    const int tid  = threadIdx.x;
    const int wave = tid >> 5;
    const int lane = tid & 31;
    const int hi   = (lane >= 16) ? 1 : 0;
    const int l16  = lane & 15;
    const int colblock = blockIdx.x * 64;

    // ---- stage activations into LDS (16B chunks, ASYNCcnt) ----
    constexpr int XCH = 64 * (Kx / 8);   // 16B chunks in x tile
    constexpr int HCH = 64 * (HH / 8);
    for (int c = tid; c < XCH; c += 256) {
        const int r = c / (Kx / 8), q = c % (Kx / 8);
        async_cp16(&sx[r * KP + q * 8], x + (size_t)r * Kx + q * 8);
    }
    for (int c = tid; c < HCH; c += 256) {
        const int r = c / (HH / 8), q = c % (HH / 8);
        async_cp16(&shm[r * HP + q * 8], h + (size_t)r * HH + q * 8);
    }
    wait_async0();
    __syncthreads();

    for (int t = wave; t < 16; t += 8) {
        const int rt   = t >> 2, ct = t & 3;
        const int row0 = rt * 16;                 // batch rows
        const int col0 = colblock + ct * 16;      // hidden cols
        const int arow = row0 + l16;              // activation row this lane
        const int wrow = col0 + l16;              // weight row this lane

        v8f ar  = {};   // r gate (gi+gh fused)
        v8f az  = {};   // z gate (gi+gh fused)
        v8f ain = {};   // n gate, input part
        v8f ahn = {};   // n gate, hidden part

        // gi = x @ Wih.T   (A from LDS, B streamed from global)
        for (int k = 0; k < Kx; k += 32) {
            v16bf xa = load_frag(sx,  arow,          KP, k, hi);
            v16bf wr = load_frag(Wih, wrow,          Kx, k, hi);
            v16bf wz = load_frag(Wih, HH + wrow,     Kx, k, hi);
            v16bf wn = load_frag(Wih, 2 * HH + wrow, Kx, k, hi);
            ar  = __builtin_amdgcn_wmma_f32_16x16x32_bf16(false, xa, false, wr, (short)0, ar,  false, false);
            az  = __builtin_amdgcn_wmma_f32_16x16x32_bf16(false, xa, false, wz, (short)0, az,  false, false);
            ain = __builtin_amdgcn_wmma_f32_16x16x32_bf16(false, xa, false, wn, (short)0, ain, false, false);
        }
        // gh = h @ Whh.T
        for (int k = 0; k < HH; k += 32) {
            v16bf ha = load_frag(shm, arow,          HP, k, hi);
            v16bf ur = load_frag(Whh, wrow,          HH, k, hi);
            v16bf uz = load_frag(Whh, HH + wrow,     HH, k, hi);
            v16bf un = load_frag(Whh, 2 * HH + wrow, HH, k, hi);
            ar  = __builtin_amdgcn_wmma_f32_16x16x32_bf16(false, ha, false, ur, (short)0, ar,  false, false);
            az  = __builtin_amdgcn_wmma_f32_16x16x32_bf16(false, ha, false, uz, (short)0, az,  false, false);
            ahn = __builtin_amdgcn_wmma_f32_16x16x32_bf16(false, ha, false, un, (short)0, ahn, false, false);
        }

        // Gate math.  C/D layout: VGPR i, lane l -> M = i + 8*(l>=16), N = l%16.
        const int colJ = col0 + l16;
        const float br  = bih[colJ]          + bhh[colJ];
        const float bz  = bih[HH + colJ]     + bhh[HH + colJ];
        const float bin = bih[2 * HH + colJ];
        const float bhn = bhh[2 * HH + colJ];
        #pragma unroll
        for (int i = 0; i < 8; i++) {
            const int m = row0 + i + hi * 8;
            const float hp = bf2f(shm[m * HP + colJ - colblock + colblock]); // LDS read
            const float r  = sigmoidf(ar[i] + br);
            const float z  = sigmoidf(az[i] + bz);
            const float n  = tanhf(ain[i] + bin + r * (ahn[i] + bhn));
            hout[(size_t)m * HH + colJ] = f2bf((1.0f - z) * n + z * hp);
        }
    }
}

// ---------------------------------------------------------------------------
// FC: out[b, n] = h[b,:] @ fcw[n,:] + fcb[n]   (M=64, N=10000, K=1024)
// h tile staged in LDS (reused 8 waves x 2 passes x 32 k-iters).
// ---------------------------------------------------------------------------
__global__ __launch_bounds__(256) void fc_kernel(
    const u16* __restrict__ hcur,
    const u16* __restrict__ fcw, const float* __restrict__ fcb,
    float* __restrict__ out, int outLd)
{
    constexpr int HP = HH + 8;
    extern __shared__ u16 smem[];
    u16* shm = smem;              // 64 x HP

    const int tid  = threadIdx.x;
    const int wave = tid >> 5;
    const int lane = tid & 31;
    const int hi   = (lane >= 16) ? 1 : 0;
    const int l16  = lane & 15;
    const int colblock = blockIdx.x * 64;

    for (int c = tid; c < 64 * (HH / 8); c += 256) {
        const int r = c / (HH / 8), q = c % (HH / 8);
        async_cp16(&shm[r * HP + q * 8], hcur + (size_t)r * HH + q * 8);
    }
    wait_async0();
    __syncthreads();

    for (int t = wave; t < 16; t += 8) {
        const int rt   = t >> 2, ct = t & 3;
        const int row0 = rt * 16;
        const int col0 = colblock + ct * 16;
        const int arow = row0 + l16;
        int wrow = col0 + l16;
        if (wrow > VV - 1) wrow = VV - 1;          // clamp edge-tile loads

        v8f acc = {};
        for (int k = 0; k < HH; k += 32) {
            v16bf ha = load_frag(shm, arow, HP, k, hi);
            v16bf wa = load_frag(fcw, wrow, HH, k, hi);
            acc = __builtin_amdgcn_wmma_f32_16x16x32_bf16(false, ha, false, wa, (short)0, acc, false, false);
        }

        const int colJ = col0 + l16;
        if (colJ < VV) {
            const float b = fcb[colJ];
            #pragma unroll
            for (int i = 0; i < 8; i++) {
                const int m = row0 + i + hi * 8;
                out[(size_t)m * outLd + colJ] = acc[i] + b;
            }
        }
    }
}

// ---------------------------------------------------------------------------
// Argmax kernels (one block per batch row, LDS tree reduction)
// ---------------------------------------------------------------------------
__global__ __launch_bounds__(256) void argmax_f32(
    const float* __restrict__ vals, int n, int ld, int* __restrict__ tok)
{
    __shared__ float sv[256];
    __shared__ int   si[256];
    const float* p = vals + (size_t)blockIdx.x * ld;
    float best = -3.4e38f; int bi = 0;
    for (int i = threadIdx.x; i < n; i += 256) {
        float v = p[i];
        if (v > best || (v == best && i < bi)) { best = v; bi = i; }
    }
    sv[threadIdx.x] = best; si[threadIdx.x] = bi;
    __syncthreads();
    for (int s = 128; s > 0; s >>= 1) {
        if (threadIdx.x < s) {
            float v2 = sv[threadIdx.x + s]; int i2 = si[threadIdx.x + s];
            if (v2 > sv[threadIdx.x] ||
                (v2 == sv[threadIdx.x] && i2 < si[threadIdx.x])) {
                sv[threadIdx.x] = v2; si[threadIdx.x] = i2;
            }
        }
        __syncthreads();
    }
    if (threadIdx.x == 0) tok[blockIdx.x] = si[0];
}

__global__ __launch_bounds__(256) void argmax_bf16(
    const u16* __restrict__ vals, int n, int ld, int* __restrict__ tok)
{
    __shared__ float sv[256];
    __shared__ int   si[256];
    const u16* p = vals + (size_t)blockIdx.x * ld;
    float best = -3.4e38f; int bi = 0;
    for (int i = threadIdx.x; i < n; i += 256) {
        float v = bf2f(p[i]);
        if (v > best || (v == best && i < bi)) { best = v; bi = i; }
    }
    sv[threadIdx.x] = best; si[threadIdx.x] = bi;
    __syncthreads();
    for (int s = 128; s > 0; s >>= 1) {
        if (threadIdx.x < s) {
            float v2 = sv[threadIdx.x + s]; int i2 = si[threadIdx.x + s];
            if (v2 > sv[threadIdx.x] ||
                (v2 == sv[threadIdx.x] && i2 < si[threadIdx.x])) {
                sv[threadIdx.x] = v2; si[threadIdx.x] = i2;
            }
        }
        __syncthreads();
    }
    if (threadIdx.x == 0) tok[blockIdx.x] = si[0];
}

// ---------------------------------------------------------------------------
// Gathers + converts + zero-init
// ---------------------------------------------------------------------------
__global__ void gather_emb(const u16* __restrict__ embbf,
                           const int* __restrict__ tok,
                           u16* __restrict__ xout)
{
    const int b = blockIdx.x;
    const int id = tok[b];
    const u16* src = embbf + (size_t)id * EE;
    u16* dst = xout + (size_t)b * EE;
    for (int e = threadIdx.x; e < EE; e += blockDim.x) dst[e] = src[e];
}

// seq layout (T, B, E): row = t*B + b so each timestep is a contiguous 64xE block
__global__ void gather_seq(const u16* __restrict__ embbf,
                           const int* __restrict__ ids,  // (B, T)
                           u16* __restrict__ xseq)
{
    const int bt = blockIdx.x;           // b*T + t
    const int b = bt / TT, t = bt % TT;
    const int id = ids[bt];
    const u16* src = embbf + (size_t)id * EE;
    u16* dst = xseq + ((size_t)t * BB + b) * EE;
    for (int e = threadIdx.x; e < EE; e += blockDim.x) dst[e] = src[e];
}

__global__ void cvt_bf16(const float* __restrict__ src, u16* __restrict__ dst, int n)
{
    const int i = blockIdx.x * blockDim.x + threadIdx.x;
    if (i < n) dst[i] = f2bf(src[i]);
}

__global__ void zero_bf16(u16* __restrict__ dst, int n)
{
    const int i = blockIdx.x * blockDim.x + threadIdx.x;
    if (i < n) dst[i] = 0;
}

// ---------------------------------------------------------------------------
// Host orchestration (all launches on `stream`; graph-capture safe)
// ---------------------------------------------------------------------------
extern "C" void kernel_launch(void* const* d_in, const int* in_sizes, int n_in,
                              void* d_out, int out_size, void* d_ws, size_t ws_size,
                              hipStream_t stream)
{
    const int*   x_ids = (const int*)  d_in[0];
    const float* emb   = (const float*)d_in[1];
    const float* wih0  = (const float*)d_in[2];
    const float* whh0  = (const float*)d_in[3];
    const float* bih0  = (const float*)d_in[4];
    const float* bhh0  = (const float*)d_in[5];
    const float* wih1  = (const float*)d_in[6];
    const float* whh1  = (const float*)d_in[7];
    const float* bih1  = (const float*)d_in[8];
    const float* bhh1  = (const float*)d_in[9];
    const float* fcw   = (const float*)d_in[10];
    const float* fcb   = (const float*)d_in[11];
    float* preds = (float*)d_out;        // (B=64, 180, V=10000) f32

    char* ws = (char*)d_ws;
    size_t off = 0;
    auto alloc = [&](size_t bytes) -> char* {
        char* p = ws + off;
        off += (bytes + 255) & ~(size_t)255;
        return p;
    };
    u16* emb_bf  = (u16*)alloc((size_t)VV * EE * 2);
    u16* wih0_bf = (u16*)alloc((size_t)3 * HH * EE * 2);
    u16* whh0_bf = (u16*)alloc((size_t)3 * HH * HH * 2);
    u16* wih1_bf = (u16*)alloc((size_t)3 * HH * HH * 2);
    u16* whh1_bf = (u16*)alloc((size_t)3 * HH * HH * 2);
    u16* fcw_bf  = (u16*)alloc((size_t)VV * HH * 2);
    u16* xseq    = (u16*)alloc((size_t)TT * BB * EE * 2);
    u16* h0seq   = (u16*)alloc((size_t)TT * BB * HH * 2);
    u16* h1seq   = (u16*)alloc((size_t)TT * BB * HH * 2);
    u16* hzero   = (u16*)alloc((size_t)BB * HH * 2);
    u16* h0bufA  = (u16*)alloc((size_t)BB * HH * 2);
    u16* h0bufB  = (u16*)alloc((size_t)BB * HH * 2);
    u16* h1bufA  = (u16*)alloc((size_t)BB * HH * 2);
    u16* h1bufB  = (u16*)alloc((size_t)BB * HH * 2);
    u16* xdec    = (u16*)alloc((size_t)BB * EE * 2);
    int* tok     = (int*)alloc((size_t)BB * 4);

    auto cvt = [&](const float* s, u16* d, int n) {
        cvt_bf16<<<(n + 255) / 256, 256, 0, stream>>>(s, d, n);
    };
    cvt(emb,  emb_bf,  VV * EE);
    cvt(wih0, wih0_bf, 3 * HH * EE);
    cvt(whh0, whh0_bf, 3 * HH * HH);
    cvt(wih1, wih1_bf, 3 * HH * HH);
    cvt(whh1, whh1_bf, 3 * HH * HH);
    cvt(fcw,  fcw_bf,  VV * HH);
    zero_bf16<<<(BB * HH + 255) / 256, 256, 0, stream>>>(hzero, BB * HH);

    // Embed entire input sequence once.
    gather_seq<<<BB * TT, 128, 0, stream>>>(emb_bf, x_ids, xseq);

    // Dynamic LDS sizes (rows padded +8 bf16 elems)
    const size_t shGru0 = (size_t)(64 * (EE + 8) + 64 * (HH + 8)) * 2; // ~199 KB
    const size_t shGru1 = (size_t)(64 * (HH + 8) + 64 * (HH + 8)) * 2; // ~264 KB
    const size_t shFc   = (size_t)(64 * (HH + 8)) * 2;                 // ~132 KB

    // Encoder, layer 0 then layer 1 (layer 1 consumes the full h0 sequence).
    for (int t = 0; t < TT; t++) {
        const u16* hp = (t == 0) ? hzero : h0seq + (size_t)(t - 1) * BB * HH;
        gru_step<EE><<<HH / 64, 256, shGru0, stream>>>(
            xseq + (size_t)t * BB * EE, hp,
            wih0_bf, whh0_bf, bih0, bhh0,
            h0seq + (size_t)t * BB * HH);
    }
    for (int t = 0; t < TT; t++) {
        const u16* hp = (t == 0) ? hzero : h1seq + (size_t)(t - 1) * BB * HH;
        gru_step<HH><<<HH / 64, 256, shGru1, stream>>>(
            h0seq + (size_t)t * BB * HH, hp,
            wih1_bf, whh1_bf, bih1, bhh1,
            h1seq + (size_t)t * BB * HH);
    }

    const u16* last = h1seq + (size_t)(TT - 1) * BB * HH;
    const int OUTLD = (DEC + 1) * VV;   // 180 * 10000 row stride in preds

    // pred0 = last @ fcw.T + fcb ; tok0 = argmax(last) over H (per reference!)
    fc_kernel<<<(VV + 63) / 64, 256, shFc, stream>>>(last, fcw_bf, fcb, preds, OUTLD);
    argmax_bf16<<<BB, 256, 0, stream>>>(last, HH, HH, tok);
    gather_emb<<<BB, 128, 0, stream>>>(emb_bf, tok, xdec);

    // Decoder: double-buffer hidden states; xdec single-buffered (in-order stream).
    const u16* h0cur = h0seq + (size_t)(TT - 1) * BB * HH;
    const u16* h1cur = last;
    u16* h0b[2] = { h0bufA, h0bufB };
    u16* h1b[2] = { h1bufA, h1bufB };
    for (int s = 0; s < DEC; s++) {
        u16* h0n = h0b[s & 1];
        u16* h1n = h1b[s & 1];
        gru_step<EE><<<HH / 64, 256, shGru0, stream>>>(xdec, h0cur,
                                                       wih0_bf, whh0_bf, bih0, bhh0, h0n);
        gru_step<HH><<<HH / 64, 256, shGru1, stream>>>(h0n, h1cur,
                                                       wih1_bf, whh1_bf, bih1, bhh1, h1n);
        float* logits = preds + (size_t)(s + 1) * VV;
        fc_kernel<<<(VV + 63) / 64, 256, shFc, stream>>>(h1n, fcw_bf, fcb, logits, OUTLD);
        argmax_f32<<<BB, 256, 0, stream>>>(logits, VV, OUTLD, tok);
        gather_emb<<<BB, 128, 0, stream>>>(emb_bf, tok, xdec);
        h0cur = h0n; h1cur = h1n;
    }
}
